// Int4Linear_90752658964992
// MI455X (gfx1250) — matrix operand order repaired and checked by reading, not compiled, not run on hardware
//
#include <hip/hip_runtime.h>

typedef __attribute__((ext_vector_type(16))) _Float16 v16h;
typedef __attribute__((ext_vector_type(8)))  _Float16 v8h;
typedef __attribute__((ext_vector_type(4)))  _Float16 v4h;
typedef __attribute__((ext_vector_type(8)))  float    v8f;
typedef __attribute__((ext_vector_type(4)))  float    v4f;
typedef __attribute__((ext_vector_type(4)))  int      v4i;

namespace {
constexpr int M_DIM = 4096;    // B*S tokens
constexpr int K_DIM = 4096;    // in_features
constexpr int N_DIM = 11008;   // out_features
constexpr int GRP   = 64;      // quant group size
constexpr int BM = 128, BN = 256, BK = 32;
constexpr int LDK = BK + 8;    // padded LDS row (halves) to spread banks
constexpr int NTHREADS = 512;  // 16 waves (wave32), 2x8 wave grid
constexpr int A_PASS = (BM * (BK / 4)) / NTHREADS;  // 2: each thread 2x float4
constexpr int W_PASS = (BN * (BK / 8)) / NTHREADS;  // 2: each thread 2x int4
}

__global__ __launch_bounds__(NTHREADS)
void int4linear_wmma_f16(const float* __restrict__ x,
                         const int*   __restrict__ packed,
                         const float* __restrict__ scales,
                         const int*   __restrict__ zeros,
                         const float* __restrict__ bias,
                         float* __restrict__ out)
{
    __shared__ _Float16 As[2][BM][LDK];   // activations, f16   (20 KB)
    __shared__ _Float16 Bs[2][BN][LDK];   // dequant weights, f16 (40 KB)

    const int tid   = threadIdx.x;
    const int lane  = tid & 31;
    const int wave  = tid >> 5;
    const int waveM = wave >> 3;   // 0..1  (64-row band)
    const int waveN = wave & 7;    // 0..7  (32-col band)

    const int mBase = blockIdx.y * BM;
    const int nBase = blockIdx.x * BN;

    // A loader: 8 threads per row (float4 each), 64 rows/pass, 2 passes
    const int aRow = tid >> 3;           // 0..63
    const int aCol = (tid & 7) * 4;
    // W loader: 4 threads per row (int4 -> 8 nibbles), 128 rows/pass, 2 passes
    const int wRow = tid >> 2;           // 0..127
    const int wJ   = (tid & 3) * 4;      // int32 index within the 16-int row slice

    v8f acc[4][2];
#pragma unroll
    for (int t = 0; t < 4; ++t)
#pragma unroll
        for (int tn = 0; tn < 2; ++tn)
#pragma unroll
            for (int r = 0; r < 8; ++r) acc[t][tn][r] = 0.0f;

    v4f   aReg[A_PASS];
    v4i   wReg[W_PASS];
    float sReg[W_PASS];
    float zReg[W_PASS];

    auto fetch = [&](int k0) {
#pragma unroll
        for (int p = 0; p < A_PASS; ++p) {
            const int row = aRow + 64 * p;
            aReg[p] = *(const v4f*)(x + (size_t)(mBase + row) * K_DIM + k0 + aCol);
        }
        const int g = k0 / GRP;   // BK=32 divides GRP=64: one group per K-step
#pragma unroll
        for (int p = 0; p < W_PASS; ++p) {
            const int    row = wRow + 128 * p;
            const size_t o   = (size_t)(nBase + row);
            wReg[p] = *(const v4i*)(packed + o * (K_DIM / 2) + (k0 >> 1) + wJ);
            sReg[p] = scales[o * (K_DIM / GRP) + g];
            zReg[p] = (float)zeros[o * (K_DIM / GRP) + g];
        }
    };

    auto stage = [&](int buf) {
#pragma unroll
        for (int p = 0; p < A_PASS; ++p) {
            const int row = aRow + 64 * p;
            v4h h;
#pragma unroll
            for (int i = 0; i < 4; ++i) h[i] = (_Float16)aReg[p][i];
            *(v4h*)&As[buf][row][aCol] = h;                    // ds_store_b64
        }
#pragma unroll
        for (int p = 0; p < W_PASS; ++p) {
            const int   row = wRow + 128 * p;
            const float s = sReg[p], z = zReg[p];
            v8h h;
#pragma unroll
            for (int i = 0; i < 4; ++i) {
                const int w = wReg[p][i];                      // nibbles in bits [7:0]
                h[2 * i]     = (_Float16)(((float)(w & 15)        - z) * s); // even k
                h[2 * i + 1] = (_Float16)(((float)((w >> 4) & 15) - z) * s); // odd k
            }
            *(v8h*)&Bs[buf][row][wJ * 2] = h;                  // ds_store_b128
        }
    };

    fetch(0);
    stage(0);

    for (int k0 = 0; k0 < K_DIM; k0 += BK) {
        const int  buf  = (k0 / BK) & 1;
        const bool more = (k0 + BK) < K_DIM;
        __syncthreads();
        if (more) fetch(k0 + BK);   // global prefetch overlaps WMMA below

        // B fragments: column n = lane&15; two 8-half K chunks per ISA layout
        v16h bfrag[2];
#pragma unroll
        for (int tn = 0; tn < 2; ++tn) {
            const _Float16* p =
                &Bs[buf][waveN * 32 + tn * 16 + (lane & 15)][(lane >> 4) * 8];
            v8h lo = *(const v8h*)p;
            v8h hi = *(const v8h*)(p + 16);
            bfrag[tn] = __builtin_shufflevector(lo, hi,
                0, 1, 2, 3, 4, 5, 6, 7, 8, 9, 10, 11, 12, 13, 14, 15);
        }
#pragma unroll
        for (int t = 0; t < 4; ++t) {
            // A fragment: row m = lane&15; two 8-half K chunks per ISA layout
            const _Float16* p =
                &As[buf][waveM * 64 + t * 16 + (lane & 15)][(lane >> 4) * 8];
            v8h lo = *(const v8h*)p;
            v8h hi = *(const v8h*)(p + 16);
            v16h afrag = __builtin_shufflevector(lo, hi,
                0, 1, 2, 3, 4, 5, 6, 7, 8, 9, 10, 11, 12, 13, 14, 15);
#pragma unroll
            for (int tn = 0; tn < 2; ++tn) {
                acc[t][tn] = __builtin_amdgcn_wmma_f32_16x16x32_f16(
                    /*neg_a=*/false, afrag, /*neg_b=*/false, bfrag[tn],
                    /*c_mod=*/(short)0, acc[t][tn],
                    /*reuse_a=*/false, /*reuse_b=*/false);
            }
        }
        if (more) stage(buf ^ 1);   // write other buffer; safe past the barrier
    }

    // Epilogue: C/D layout -> VGPR r holds M = r + 8*(lane>=16); N = lane&15
#pragma unroll
    for (int t = 0; t < 4; ++t) {
        const int m0 = mBase + waveM * 64 + t * 16 + (lane >> 4) * 8;
#pragma unroll
        for (int tn = 0; tn < 2; ++tn) {
            const int   n  = nBase + waveN * 32 + tn * 16 + (lane & 15);
            const float bv = bias[n];
#pragma unroll
            for (int r = 0; r < 8; ++r) {
                out[(size_t)(m0 + r) * N_DIM + n] = acc[t][tn][r] + bv;
            }
        }
    }
}

extern "C" void kernel_launch(void* const* d_in, const int* in_sizes, int n_in,
                              void* d_out, int out_size, void* d_ws, size_t ws_size,
                              hipStream_t stream) {
    const float* x      = (const float*)d_in[0];
    const int*   packed = (const int*)d_in[1];
    const float* scales = (const float*)d_in[2];
    const int*   zeros  = (const int*)d_in[3];
    const float* bias   = (const float*)d_in[4];
    float*       out    = (float*)d_out;

    dim3 grid(N_DIM / BN, M_DIM / BM);  // (43, 32)
    int4linear_wmma_f16<<<grid, NTHREADS, 0, stream>>>(x, packed, scales, zeros, bias, out);
}
